// BasicGCNRegressor_23089744183692
// MI455X (gfx1250) — matrix operand order repaired and checked by reading, not compiled, not run on hardware
//
#include <hip/hip_runtime.h>
#include <hip/hip_bf16.h>

#define N_NODES   100000
#define N_EDGES   1600000
#define DIM       128
#define N_CLASSES 4

typedef __attribute__((ext_vector_type(16))) __bf16 v16bf;
typedef __attribute__((ext_vector_type(8)))  float  v8f;

// ---------------------------------------------------------------- utilities
__global__ __launch_bounds__(256) void k_zero(float* __restrict__ p, size_t n) {
  size_t i = (size_t)blockIdx.x * blockDim.x + threadIdx.x;
  size_t stride = (size_t)gridDim.x * blockDim.x;
  for (; i < n; i += stride) p[i] = 0.0f;
}

// ---------------------------------------------------------------- degrees
__global__ __launch_bounds__(256) void k_degree(const int* __restrict__ src,
                                                const int* __restrict__ dst,
                                                float* __restrict__ outdeg,
                                                float* __restrict__ indeg) {
  int e = blockIdx.x * blockDim.x + threadIdx.x;
  if (e < N_EDGES) {
    atomicAdd(&outdeg[src[e]], 1.0f);
    atomicAdd(&indeg[dst[e]], 1.0f);
  }
}

__global__ __launch_bounds__(256) void k_norm(const float* __restrict__ outdeg,
                                              const float* __restrict__ indeg,
                                              float* __restrict__ ns,
                                              float* __restrict__ nd) {
  int i = blockIdx.x * blockDim.x + threadIdx.x;
  if (i < N_NODES) {
    float o = outdeg[i], d = indeg[i];
    ns[i] = (o > 0.0f) ? rsqrtf(fmaxf(o, 1.0f)) : 0.0f;
    nd[i] = (d > 0.0f) ? rsqrtf(fmaxf(d, 1.0f)) : 0.0f;
  }
}

// ---------------------------------------------------------------- edge scatter
// one wave per edge; lane handles 4 features (float4 gather, 4 f32 atomics)
__global__ __launch_bounds__(256) void k_scatter(const float* __restrict__ x,
                                                 const int* __restrict__ src,
                                                 const int* __restrict__ dst,
                                                 const float* __restrict__ ns,
                                                 float* __restrict__ agg) {
  int wave = threadIdx.x >> 5;
  int lane = threadIdx.x & 31;
  int e = blockIdx.x * 8 + wave;
  if (e >= N_EDGES) return;
  int s = src[e], d = dst[e];
  float w = ns[s];
  const float4* xv = (const float4*)(x + (size_t)s * DIM);
  float4 v = xv[lane];
  float* ap = agg + (size_t)d * DIM + lane * 4;
  atomicAdd(ap + 0, v.x * w);
  atomicAdd(ap + 1, v.y * w);
  atomicAdd(ap + 2, v.z * w);
  atomicAdd(ap + 3, v.w * w);
}

// ---------------------------------------------------------------- WMMA GEMM
// Out = relu( diag(norm_dst) * (A @ W) + bias ), A:[N,128] W:[128,128]
// 8 waves/block; wave w owns output columns 16w..16w+15 for every row tile.
// B fragments (bf16) preloaded once per wave: lane&15 = column within stripe,
// element i = W[k][col0+(lane&15)], k = (i&7)+8*(lane>>4)+16*(i>>3) (dense 32x16 B layout, ISA 7.12.2).
// A fragment: row = lane&15, K halves selected by lane>>4 (16-bit A layout).
// Epilogue uses hoisted base pointers so the unrolled M index becomes an
// immediate VMEM offset (i*512B) instead of per-element 64-bit address math.
__global__ __launch_bounds__(256) void k_gemm_relu(const float* __restrict__ A,
                                                   const float* __restrict__ W,
                                                   const float* __restrict__ bias,
                                                   const float* __restrict__ norm_dst,
                                                   float* __restrict__ Out,
                                                   int n_tiles) {
  const int wave = threadIdx.x >> 5;
  const int lane = threadIdx.x & 31;
  const int col0 = wave * 16;

  // preload B stripe: 4 K-blocks of 32, 16 bf16 per lane each
  v16bf b[4];
#pragma unroll
  for (int kb = 0; kb < 4; ++kb) {
    const float* wrow = W + (size_t)(kb * 32) * DIM + col0 + (lane & 15);
#pragma unroll
    for (int i = 0; i < 16; ++i) b[kb][i] = (__bf16)wrow[(size_t)(((i & 7) + 8 * (lane >> 4) + 16 * (i >> 3)) * DIM)];
  }

  const int row_in_tile = lane & 15;
  const int half = lane >> 4;
  const int no = col0 + (lane & 15);
  const float bv = bias[no];

  for (int t = blockIdx.x; t < n_tiles; t += gridDim.x) {
    const float* arow = A + (size_t)(t * 16 + row_in_tile) * DIM;
    v8f c = {};
#pragma unroll
    for (int kb = 0; kb < 4; ++kb) {
      v16bf a;
      const float* p0 = arow + kb * 32 + half * 8;
      const float* p1 = p0 + 16;
#pragma unroll
      for (int i = 0; i < 8; ++i) {
        a[i]     = (__bf16)p0[i];
        a[8 + i] = (__bf16)p1[i];
      }
      c = __builtin_amdgcn_wmma_f32_16x16x32_bf16(
          /*neg_a=*/false, a, /*neg_b=*/false, b[kb],
          /*c_mod=*/(short)0, c, /*reuse_a=*/false, /*reuse_b=*/false);
    }
    // C/D layout: lane -> N = lane&15, VGPR i -> M = i + 8*(lane>>4).
    // One base per tile; i becomes an immediate offset (i*DIM floats).
    const int m0 = t * 16 + 8 * half;
    const float* ndp = norm_dst + m0;
    float* op = Out + (size_t)m0 * DIM + no;
#pragma unroll
    for (int i = 0; i < 8; ++i) {
      float v = ndp[i] * c[i] + bv;
      op[i * DIM] = v > 0.0f ? v : 0.0f;
    }
  }
}

// ---------------------------------------------------------------- column sum
__global__ __launch_bounds__(128) void k_colsum(const float* __restrict__ h,
                                                float* __restrict__ acc) {
  int j = threadIdx.x;
  int n0 = blockIdx.x * 256;
  int n1 = n0 + 256; if (n1 > N_NODES) n1 = N_NODES;
  float s = 0.0f;
  for (int n = n0; n < n1; ++n) s += h[(size_t)n * DIM + j];
  atomicAdd(&acc[j], s);
}

__global__ __launch_bounds__(32) void k_seg(const float* __restrict__ acc,
                                            const float* __restrict__ Wp,
                                            const float* __restrict__ bp,
                                            float* __restrict__ out) {
  int c = threadIdx.x;
  if (c < N_CLASSES) {
    float s = 0.0f;
    for (int j = 0; j < DIM; ++j) s += Wp[c * DIM + j] * acc[j];
    out[c] = s * (1.0f / (float)N_NODES) + bp[c];
  }
}

// ---------------------------------------------------------------- CAM: Wp @ h^T
// Output is write-once / never re-read -> non-temporal stores so it doesn't
// evict the L2-resident node buffers.
__global__ __launch_bounds__(256) void k_cam(const float* __restrict__ h,
                                             const float* __restrict__ Wp,
                                             float* __restrict__ cam) {
  __shared__ float wp[N_CLASSES * DIM];
  for (int i = threadIdx.x; i < N_CLASSES * DIM; i += blockDim.x) wp[i] = Wp[i];
  __syncthreads();
  int n = blockIdx.x * blockDim.x + threadIdx.x;
  if (n >= N_NODES) return;
  const float4* hv = (const float4*)(h + (size_t)n * DIM);
  float a0 = 0.f, a1 = 0.f, a2 = 0.f, a3 = 0.f;
#pragma unroll 4
  for (int k = 0; k < DIM / 4; ++k) {
    float4 v = hv[k];
    int kk = k * 4;
    a0 += v.x * wp[0 * DIM + kk] + v.y * wp[0 * DIM + kk + 1] + v.z * wp[0 * DIM + kk + 2] + v.w * wp[0 * DIM + kk + 3];
    a1 += v.x * wp[1 * DIM + kk] + v.y * wp[1 * DIM + kk + 1] + v.z * wp[1 * DIM + kk + 2] + v.w * wp[1 * DIM + kk + 3];
    a2 += v.x * wp[2 * DIM + kk] + v.y * wp[2 * DIM + kk + 1] + v.z * wp[2 * DIM + kk + 2] + v.w * wp[2 * DIM + kk + 3];
    a3 += v.x * wp[3 * DIM + kk] + v.y * wp[3 * DIM + kk + 1] + v.z * wp[3 * DIM + kk + 2] + v.w * wp[3 * DIM + kk + 3];
  }
  __builtin_nontemporal_store(a0, cam + 0 * (size_t)N_NODES + n);
  __builtin_nontemporal_store(a1, cam + 1 * (size_t)N_NODES + n);
  __builtin_nontemporal_store(a2, cam + 2 * (size_t)N_NODES + n);
  __builtin_nontemporal_store(a3, cam + 3 * (size_t)N_NODES + n);
}

// ---------------------------------------------------------------- launcher
extern "C" void kernel_launch(void* const* d_in, const int* in_sizes, int n_in,
                              void* d_out, int out_size, void* d_ws, size_t ws_size,
                              hipStream_t stream) {
  const float* features = (const float*)d_in[0];
  const int*   src      = (const int*)d_in[1];
  const int*   dst      = (const int*)d_in[2];
  // d_in[3] = is_training (unused)
  const float* W1 = (const float*)d_in[4];
  const float* b1 = (const float*)d_in[5];
  const float* W2 = (const float*)d_in[6];
  const float* b2 = (const float*)d_in[7];
  const float* Wp = (const float*)d_in[8];
  const float* bp = (const float*)d_in[9];
  float* out = (float*)d_out;

  const size_t NF = (size_t)N_NODES * DIM;
  float* ws     = (float*)d_ws;
  float* bufA   = ws;               // aggregation buffer [N,128]
  float* bufB   = bufA + NF;        // hidden buffer      [N,128]
  float* outdeg = bufB + NF;        // [N]
  float* indeg  = outdeg + N_NODES; // [N]
  float* hg     = indeg + N_NODES;  // [128] column-sum accumulator
  float* ns     = hg + DIM;         // [N]
  float* nd     = ns + N_NODES;     // [N]

  const int Z_BLOCKS    = 2048;
  const int EDGE_BLOCKS = (N_EDGES + 7) / 8;          // 1 wave per edge
  const int DEG_BLOCKS  = (N_EDGES + 255) / 256;
  const int NODE_BLOCKS = (N_NODES + 255) / 256;
  const int TILES       = N_NODES / 16;               // 6250, exact
  const int GEMM_BLOCKS = 1024;

  // zero: agg buffer + [outdeg, indeg, hg] (contiguous)
  k_zero<<<Z_BLOCKS, 256, 0, stream>>>(bufA, NF);
  k_zero<<<Z_BLOCKS, 256, 0, stream>>>(outdeg, 2 * (size_t)N_NODES + DIM);

  // degrees + norms
  k_degree<<<DEG_BLOCKS, 256, 0, stream>>>(src, dst, outdeg, indeg);
  k_norm<<<NODE_BLOCKS, 256, 0, stream>>>(outdeg, indeg, ns, nd);

  // layer 1: scatter(features) -> bufA; gemm -> bufB
  k_scatter<<<EDGE_BLOCKS, 256, 0, stream>>>(features, src, dst, ns, bufA);
  k_gemm_relu<<<GEMM_BLOCKS, 256, 0, stream>>>(bufA, W1, b1, nd, bufB, TILES);

  // layer 2: re-zero bufA; scatter(bufB) -> bufA; gemm -> bufB
  k_zero<<<Z_BLOCKS, 256, 0, stream>>>(bufA, NF);
  k_scatter<<<EDGE_BLOCKS, 256, 0, stream>>>(bufB, src, dst, ns, bufA);
  k_gemm_relu<<<GEMM_BLOCKS, 256, 0, stream>>>(bufA, W2, b2, nd, bufB, TILES);

  // head: mean-pool -> seg_output [4], then CAM [4, N]
  k_colsum<<<NODE_BLOCKS, 128, 0, stream>>>(bufB, hg);
  k_seg<<<1, 32, 0, stream>>>(hg, Wp, bp, out);
  k_cam<<<NODE_BLOCKS, 256, 0, stream>>>(bufB, Wp, out + N_CLASSES);
}
